// PolicyValue_35321811042359
// MI455X (gfx1250) — compile-verified, matrix operand order checked by reading, not run on hardware
//
#include <hip/hip_runtime.h>
#include <hip/hip_bf16.h>
#include <climits>

typedef __attribute__((ext_vector_type(16))) _Float16 v16h;
typedef __attribute__((ext_vector_type(8)))  float    v8f;

__device__ __forceinline__ long lower_bound_i32(const int* __restrict__ a, long n, int key) {
    long lo = 0, hi = n;
    while (lo < hi) {
        long mid = (lo + hi) >> 1;
        if (a[mid] < key) lo = mid + 1; else hi = mid;
    }
    return lo;
}

// =====================================================================
// K1: one block per graph (indices are sorted => contiguous segments).
// Computes: segment max, denom = sum(exp(pi-max)), log_pi for its nodes,
// and the 64-channel segment-sum of value_embeds -> parts[g].
// No global atomics anywhere; value_embeds streamed once with b128 loads.
// =====================================================================
__global__ __launch_bounds__(256)
void k_graph(const float* __restrict__ pi,
             const float* __restrict__ ve,
             const int*   __restrict__ gidx,
             float* __restrict__ log_pi,
             float* __restrict__ parts,
             long N, int G) {
    __shared__ long   seS[2];
    __shared__ float  red[256];
    __shared__ float4 red4[256];

    const int g = blockIdx.x;
    const int t = threadIdx.x;

    if (t < 2) seS[t] = lower_bound_i32(gidx, N, g + t);
    __syncthreads();
    const long s = seS[0];
    const long e = seS[1];

    // ---- pass 1: segment max ----
    float m = -INFINITY;
    for (long i = s + t; i < e; i += 256) m = fmaxf(m, pi[i]);
    red[t] = m;
    __syncthreads();
    for (int off = 128; off > 0; off >>= 1) {
        if (t < off) red[t] = fmaxf(red[t], red[t + off]);
        __syncthreads();
    }
    const float gmax = red[0];
    __syncthreads();

    // ---- pass 2: denom = sum(exp(pi - max)) ----
    float sacc = 0.0f;
    for (long i = s + t; i < e; i += 256) sacc += expf(pi[i] - gmax);
    red[t] = sacc;
    __syncthreads();
    for (int off = 128; off > 0; off >>= 1) {
        if (t < off) red[t] = red[t] + red[t + off];
        __syncthreads();
    }
    const float ldenom = logf(red[0]);
    __syncthreads();

    // ---- pass 3: log_pi = pi - max - log(denom) ----
    for (long i = s + t; i < e; i += 256) log_pi[i] = pi[i] - gmax - ldenom;

    // ---- pass 4: parts[g, :] = segment-sum of value_embeds rows ----
    // 16 node-stripes x 16 float4 channel-groups; b128 coalesced streaming.
    const int c4 = t & 15;      // float4 group within the 64-channel row
    const int q  = t >> 4;      // node stripe 0..15
    const float4* __restrict__ ve4 = (const float4*)ve;
    float4 acc = make_float4(0.f, 0.f, 0.f, 0.f);
    for (long i = s + q; i < e; i += 16) {
        float4 v = ve4[i * 16 + c4];
        acc.x += v.x; acc.y += v.y; acc.z += v.z; acc.w += v.w;
    }
    red4[t] = acc;              // t == q*16 + c4
    __syncthreads();
    if (t < 16) {
        float4 sum = red4[t];
        #pragma unroll
        for (int qq = 1; qq < 16; ++qq) {
            float4 v = red4[qq * 16 + t];
            sum.x += v.x; sum.y += v.y; sum.z += v.z; sum.w += v.w;
        }
        ((float4*)parts)[(long)g * 16 + t] = sum;
    }
}

// =====================================================================
// K2: value = sigmoid(parts @ W + b) on the WMMA path.
// One wave per 16 graphs; K=64 as two chained v_wmma_f32_16x16x32_f16.
// A-fragment: contiguous 8-float runs per lane per the ISA 16-bit 16x32
// layout (K = j + 8*((j>=8)+hi)).  B carries W in column N=0 via a
// branchless lane-select multiply.  Row index clamped (no predication).
// =====================================================================
__device__ __forceinline__ v16h pack_frag(const float* __restrict__ p, int hi, float scale) {
    const float* p0 = p + 8 * hi;        // j = 0..7  -> K = j + 8*hi
    const float* p1 = p + 16 + 8 * hi;   // j = 8..15 -> K = (j-8) + 16 + 8*hi
    v16h r;
    #pragma unroll
    for (int j = 0; j < 8; ++j) {
        r[j]     = (_Float16)(p0[j] * scale);
        r[j + 8] = (_Float16)(p1[j] * scale);
    }
    return r;
}

__global__ __launch_bounds__(256)
void k_value_wmma(const float* __restrict__ parts,
                  const float* __restrict__ W,
                  const float* __restrict__ bias,
                  float* __restrict__ out_val, int G) {
    const int wave = threadIdx.x >> 5;            // 8 waves per block
    const int lane = threadIdx.x & 31;
    const int m    = lane & 15;                   // A row in tile / D column N
    const int hi   = lane >> 4;
    const long tile = (long)blockIdx.x * 8 + wave; // 16 graphs per tile

    long row = tile * 16 + m;
    if (row >= G) row = G - 1;                    // clamp: extra rows -> unsaved D rows
    const float* arow = parts + row * 64;

    const float sel = (m == 0) ? 1.0f : 0.0f;     // B column 0 = W, branchless

    v16h a0 = pack_frag(arow,      hi, 1.0f);
    v16h a1 = pack_frag(arow + 32, hi, 1.0f);
    v16h b0 = pack_frag(W,         hi, sel);
    v16h b1 = pack_frag(W + 32,    hi, sel);

    v8f c = {};
    c = __builtin_amdgcn_wmma_f32_16x16x32_f16(false, a0, false, b0, (short)0, c, false, false);
    c = __builtin_amdgcn_wmma_f32_16x16x32_f16(false, a1, false, b1, (short)0, c, false, false);

    const float bv = bias[0];
    if (m == 0) {                                  // lanes 0 & 16 hold D[:, N=0]
        #pragma unroll
        for (int r = 0; r < 8; ++r) {
            long gOut = tile * 16 + r + 8 * hi;    // M = r + 8*hi
            if (gOut < G) {
                float x = c[r] + bv;
                out_val[gOut] = 1.0f / (1.0f + expf(-x));
            }
        }
    }
}

extern "C" void kernel_launch(void* const* d_in, const int* in_sizes, int n_in,
                              void* d_out, int out_size, void* d_ws, size_t ws_size,
                              hipStream_t stream) {
    const float* pi   = (const float*)d_in[0];
    const float* ve   = (const float*)d_in[1];
    const float* W    = (const float*)d_in[2];
    const float* bias = (const float*)d_in[3];
    const int*   gidx = (const int*)d_in[4];

    long N = in_sizes[0];
    int  G = (int)((long)out_size - N);

    float* parts   = (float*)d_ws;       // [G,64], fully overwritten by k_graph
    float* log_pi  = (float*)d_out;      // [N]
    float* val_out = log_pi + N;         // [G]

    k_graph<<<G, 256, 0, stream>>>(pi, ve, gidx, log_pi, parts, N, G);

    int tiles  = (G + 15) / 16;
    int blocks = (tiles + 7) / 8;
    k_value_wmma<<<blocks, 256, 0, stream>>>(parts, W, bias, val_out, G);
}
